// MultiField_TransformerLayer_650
// MI455X (gfx1250) — matrix operand order repaired and checked by reading, not compiled, not run on hardware
//
#include <hip/hip_runtime.h>
#include <hip/hip_bf16.h>

// ---------------------------------------------------------------------------
// MultiField transformer layer for MI455X (gfx1250, wave32, WMMA bf16)
//   B=16, S=1024, D=128 (head dim), H=8, D*H=1024
// Pipeline:
//   0) prep:        X -> bf16; W{q,k,v,o} -> transposed bf16 (N-major)
//   1) qkv_kernel:  Q/K -> [B,H,S,D] bf16, V -> [B,H,D,S] bf16 (WMMA)
//   2) attn_kernel: flash-attention, 32 queries/wave (K/V frag reuse x2,
//                   full-wave online softmax)
//   3) oln_kernel:  out = LN(X + WV @ Wo + bo) (WMMA + LDS LayerNorm)
// All GEMM operands are contiguous bf16 -> b128 vector loads feeding WMMA.
// ---------------------------------------------------------------------------

typedef __attribute__((ext_vector_type(16))) __bf16 v16bf;
typedef __attribute__((ext_vector_type(8)))  __bf16 v8bf;
typedef __attribute__((ext_vector_type(8)))  float  v8f;

#define Bb     16
#define Ss     1024
#define Dd     128
#define Hh     8
#define DH     1024          /* D*H */
#define BS     16384         /* B*S */
#define LN_EPS 1e-5f
#define QKSCL  0.08838834764831845f  /* 1/sqrt(128) */

// Native bf16 convert (fptrunc = RNE; backend picks v_cvt_pk_bf16_f32 path)
static __device__ __forceinline__ __bf16 f2bf(float f) { return (__bf16)f; }

static __device__ __forceinline__ v8f wmma_bf16(v16bf a, v16bf b, v8f c) {
    return __builtin_amdgcn_wmma_f32_16x16x32_bf16(
        false, a, false, b, (short)0, c, false, false);
}

// ---------- vectorized fragment loaders (contiguous k = 16*(lane/16) + s, same for A and B) --------
static __device__ __forceinline__ v16bf ld_a16(const __bf16* p) {
    v8bf lo = *(const v8bf*)p;          // elements k+0..7
    v8bf hi = *(const v8bf*)(p + 8);    // elements k+8..15
    return __builtin_shufflevector(lo, hi, 0,1,2,3,4,5,6,7,8,9,10,11,12,13,14,15);
}
static __device__ __forceinline__ v16bf ld_b16(const __bf16* p) {
    v8bf lo = *(const v8bf*)p;
    v8bf hi = *(const v8bf*)(p + 8);
    return __builtin_shufflevector(lo, hi, 0,1,2,3,4,5,6,7,8,9,10,11,12,13,14,15);
}
// A[m][k] = src[m*lda + k] (16x32 tile at src)
static __device__ __forceinline__ v16bf load_a(const __bf16* src, size_t lda, int lane) {
    int r  = lane & 15;
    int kb = (lane < 16) ? 0 : 16;
    return ld_a16(src + (size_t)r * lda + kb);
}
// B[k][n] = src[n*ldn + k] (k contiguous; 32x16 tile at src)
static __device__ __forceinline__ v16bf load_b(const __bf16* src, size_t ldn, int lane) {
    int c  = lane & 15;
    int kb = (lane < 16) ? 0 : 16;
    return ld_b16(src + (size_t)c * ldn + kb);
}

// ---------------------------------------------------------------------------
// Prep kernels (one-time, bandwidth-trivial)
// ---------------------------------------------------------------------------
__global__ __launch_bounds__(256) void cvt_x_kernel(
    const float* __restrict__ X, __bf16* __restrict__ Xb) {
    int i = blockIdx.x * 256 + threadIdx.x;
    Xb[i] = f2bf(X[i]);
}
__global__ __launch_bounds__(256) void tr_w_kernel(
    const float* __restrict__ Wq, const float* __restrict__ Wk,
    const float* __restrict__ Wv, __bf16* __restrict__ Wqt,
    __bf16* __restrict__ Wkt, __bf16* __restrict__ Wvt) {
    int i = blockIdx.x * 256 + threadIdx.x;   // over DH*Dd, i = n*Dd + k
    int n = i >> 7, k = i & 127;
    const float* W = (blockIdx.y == 0) ? Wq : (blockIdx.y == 1) ? Wk : Wv;
    __bf16*     Wt = (blockIdx.y == 0) ? Wqt : (blockIdx.y == 1) ? Wkt : Wvt;
    Wt[i] = f2bf(W[(size_t)k * DH + n]);
}
__global__ __launch_bounds__(256) void tr_wo_kernel(
    const float* __restrict__ Wo, __bf16* __restrict__ Wot) {
    int i = blockIdx.x * 256 + threadIdx.x;   // over Dd*DH, i = n*DH + k
    int n = i >> 10, k = i & 1023;
    Wot[i] = f2bf(Wo[(size_t)k * Dd + n]);
}

// ---------------------------------------------------------------------------
// Kernel 1: QKV projection.  grid (BS/16, DH/128, 3), block 256 (8 waves).
// ---------------------------------------------------------------------------
__global__ __launch_bounds__(256) void qkv_kernel(
    const __bf16* __restrict__ Xb,
    const __bf16* __restrict__ Wqt, const float* __restrict__ bq,
    const __bf16* __restrict__ Wkt, const float* __restrict__ bk,
    const __bf16* __restrict__ Wvt, const float* __restrict__ bv,
    __bf16* __restrict__ Qt, __bf16* __restrict__ Kt, __bf16* __restrict__ Vt) {
    const int lane = threadIdx.x & 31;
    const int w    = threadIdx.x >> 5;
    const int m0   = blockIdx.x * 16;
    const int n0   = blockIdx.y * 128 + w * 16;
    const int which = blockIdx.z;
    const __bf16* Wt; const float* bias;
    if (which == 0)      { Wt = Wqt; bias = bq; }
    else if (which == 1) { Wt = Wkt; bias = bk; }
    else                 { Wt = Wvt; bias = bv; }

    const int c    = lane & 15;
    const int mofs = (lane < 16) ? 0 : 8;
    v8f acc = {};
#pragma unroll
    for (int kk = 0; kk < 4; ++kk) {
        const int k0 = kk * 32;
        v16bf a = load_a(Xb + (size_t)m0 * Dd + k0, Dd, lane);
        v16bf b = load_b(Wt + (size_t)n0 * Dd + k0, Dd, lane);
        acc = wmma_bf16(a, b, acc);
    }
    const float bb   = bias[n0 + c];
    const int   bi   = m0 >> 10;            // batch index
    const int   s0   = m0 & 1023;           // seq offset in batch
    const int   h    = n0 >> 7;             // head
    const int   dcol = (n0 & 127) + c;      // dim within head

    if (which < 2) {                        // Q / K -> [B,H,S,D]
        __bf16* out = (which == 0) ? Qt : Kt;
        size_t base = ((((size_t)bi * Hh + h) * Ss) + s0 + mofs) * Dd + dcol;
#pragma unroll
        for (int r = 0; r < 8; ++r) out[base + (size_t)r * Dd] = f2bf(acc[r] + bb);
    } else {                                // V -> [B,H,D,S] (contiguous rows)
        size_t base = (((size_t)bi * Hh + h) * Dd + dcol) * Ss + s0 + mofs;
        v8bf vv;
#pragma unroll
        for (int r = 0; r < 8; ++r) vv[r] = f2bf(acc[r] + bb);
        *(v8bf*)(Vt + base) = vv;
    }
}

// ---------------------------------------------------------------------------
// Kernel 2: flash attention, 32 queries per wave (2 q-tiles).
// K/V B-fragments are each reused by two WMMAs; softmax uses all 32 lanes.
// grid 512 blocks x 256 threads = 4096 waves = 16*8*(1024/32).
// ---------------------------------------------------------------------------
__global__ __launch_bounds__(256) void attn_kernel(
    const __bf16* __restrict__ Qt, const __bf16* __restrict__ Kt,
    const __bf16* __restrict__ Vt, __bf16* __restrict__ WVb) {
    __shared__ alignas(32) float  sS[8][32 * 32];   // 32 q-rows x 32 keys
    __shared__ alignas(32) __bf16 sP[8][32 * 32];
    __shared__ float sM[8][32], sL[8][32], sA[8][32];

    const int lane = threadIdx.x & 31;
    const int w    = threadIdx.x >> 5;
    const int gw   = blockIdx.x * 8 + w;
    const int qt32 = gw & 31;               // 32 query-blocks of 32
    const int h    = (gw >> 5) & 7;
    const int b    = gw >> 8;
    const int q0   = qt32 * 32;
    const int c    = lane & 15;
    const int mofs = (lane < 16) ? 0 : 8;
    const size_t bh = (size_t)b * Hh + h;
    const __bf16* Qh = Qt + bh * Ss * Dd;   // [S, D]
    const __bf16* Kh = Kt + bh * Ss * Dd;   // [S, D]
    const __bf16* Vh = Vt + bh * Dd * Ss;   // [D, S]

    sM[w][lane] = -1e30f;
    sL[w][lane] = 0.0f;
    __builtin_amdgcn_wave_barrier();
    asm volatile("s_wait_dscnt 0" ::: "memory");

    // Q fragments for both q-tiles, live across the whole key loop.
    v16bf qf[2][4];
#pragma unroll
    for (int i = 0; i < 2; ++i)
#pragma unroll
        for (int dc = 0; dc < 4; ++dc)
            qf[i][dc] = load_a(Qh + (size_t)(q0 + i * 16) * Dd + dc * 32, Dd, lane);

    v8f acc[2][8];
#pragma unroll
    for (int i = 0; i < 2; ++i)
#pragma unroll
        for (int t = 0; t < 8; ++t) acc[i][t] = (v8f){};

    for (int kt = 0; kt < 32; ++kt) {
        const int kk0 = kt * 32;
        if (kt + 1 < 32) {
            __builtin_prefetch(Kh + (size_t)(kk0 + 32 + c) * Dd, 0, 1);
            __builtin_prefetch(Vh + (size_t)(c * 16) * Ss + kk0 + 32, 0, 1);
        }

        // ----- scores: each K fragment feeds both q-tiles ------------------
        v8f s0[2] = {{}, {}}, s1[2] = {{}, {}};
#pragma unroll
        for (int dc = 0; dc < 4; ++dc) {
            v16bf kb0 = load_b(Kh + (size_t)kk0 * Dd + dc * 32, Dd, lane);
            v16bf kb1 = load_b(Kh + (size_t)(kk0 + 16) * Dd + dc * 32, Dd, lane);
#pragma unroll
            for (int i = 0; i < 2; ++i) {
                s0[i] = wmma_bf16(qf[i][dc], kb0, s0[i]);
                s1[i] = wmma_bf16(qf[i][dc], kb1, s1[i]);
            }
        }
#pragma unroll
        for (int i = 0; i < 2; ++i)
#pragma unroll
            for (int r = 0; r < 8; ++r) {
                sS[w][(i * 16 + mofs + r) * 32 + c]      = s0[i][r] * QKSCL;
                sS[w][(i * 16 + mofs + r) * 32 + 16 + c] = s1[i][r] * QKSCL;
            }
        __builtin_amdgcn_wave_barrier();
        asm volatile("s_wait_dscnt 0" ::: "memory");

        // ----- online softmax: all 32 lanes, one query row each ------------
        {
            const int r = lane;
            float mold = sM[w][r];
            float tmax = mold;
            float sv[32];
#pragma unroll
            for (int j = 0; j < 32; ++j) { sv[j] = sS[w][r * 32 + j]; tmax = fmaxf(tmax, sv[j]); }
            float alpha = __expf(mold - tmax);
            float rsum  = 0.0f;
#pragma unroll
            for (int j = 0; j < 32; ++j) {
                float p = __expf(sv[j] - tmax);
                sP[w][r * 32 + j] = f2bf(p);
                rsum += p;
            }
            sL[w][r] = sL[w][r] * alpha + rsum;
            sM[w][r] = tmax;
            sA[w][r] = alpha;
        }
        __builtin_amdgcn_wave_barrier();
        asm volatile("s_wait_dscnt 0" ::: "memory");

        // ----- rescale running accumulators by alpha(row) ------------------
#pragma unroll
        for (int i = 0; i < 2; ++i) {
            float aR[8];
#pragma unroll
            for (int r = 0; r < 8; ++r) aR[r] = sA[w][i * 16 + mofs + r];
#pragma unroll
            for (int t = 0; t < 8; ++t)
#pragma unroll
                for (int r = 0; r < 8; ++r) acc[i][t][r] *= aR[r];
        }

        // ----- O += P V : each V fragment feeds both q-tiles ---------------
        v16bf pa0 = load_a(&sP[w][0],       32, lane);
        v16bf pa1 = load_a(&sP[w][16 * 32], 32, lane);
#pragma unroll
        for (int t = 0; t < 8; ++t) {
            v16bf vb = load_b(Vh + (size_t)(t * 16) * Ss + kk0, Ss, lane);
            acc[0][t] = wmma_bf16(pa0, vb, acc[0][t]);
            acc[1][t] = wmma_bf16(pa1, vb, acc[1][t]);
        }
    }

    // ----- normalize by row sums and write WV (bf16, [BS, DH]) --------------
    __builtin_amdgcn_wave_barrier();
    asm volatile("s_wait_dscnt 0" ::: "memory");
#pragma unroll
    for (int i = 0; i < 2; ++i) {
        float linv[8];
#pragma unroll
        for (int r = 0; r < 8; ++r) linv[r] = 1.0f / sL[w][i * 16 + mofs + r];
        const size_t rowbase = (size_t)b * Ss + q0 + i * 16 + mofs;
#pragma unroll
        for (int t = 0; t < 8; ++t)
#pragma unroll
            for (int r = 0; r < 8; ++r)
                WVb[(rowbase + r) * DH + (size_t)h * Dd + t * 16 + c] =
                    f2bf(acc[i][t][r] * linv[r]);
    }
}

// ---------------------------------------------------------------------------
// Kernel 3: output projection + residual + LayerNorm.
// ---------------------------------------------------------------------------
__global__ __launch_bounds__(256) void oln_kernel(
    const __bf16* __restrict__ WVb, const __bf16* __restrict__ Wot,
    const float* __restrict__ bo,  const float* __restrict__ X,
    const float* __restrict__ gamma, const float* __restrict__ beta,
    float* __restrict__ out) {
    __shared__ alignas(32) float sT[16][128];
    __shared__ float sMean[16], sRstd[16];

    const int lane = threadIdx.x & 31;
    const int w    = threadIdx.x >> 5;
    const int m0   = blockIdx.x * 16;
    const int n0   = w * 16;
    const int c    = lane & 15;
    const int mofs = (lane < 16) ? 0 : 8;

    v8f acc = {};
#pragma unroll 4
    for (int kk = 0; kk < 32; ++kk) {
        const int k0 = kk * 32;
        v16bf a = load_a(WVb + (size_t)m0 * DH + k0, DH, lane);
        v16bf b = load_b(Wot + (size_t)n0 * DH + k0, DH, lane);
        acc = wmma_bf16(a, b, acc);
    }
    const float bb = bo[n0 + c];
#pragma unroll
    for (int r = 0; r < 8; ++r) {
        const int row = mofs + r;
        sT[row][n0 + c] = acc[r] + bb + X[(size_t)(m0 + row) * Dd + n0 + c];
    }
    __syncthreads();

    if (threadIdx.x < 16) {
        const int row = threadIdx.x;
        float s = 0.0f, s2 = 0.0f;
#pragma unroll 8
        for (int j = 0; j < 128; ++j) { float x = sT[row][j]; s += x; s2 += x * x; }
        const float mean = s * (1.0f / 128.0f);
        const float var  = s2 * (1.0f / 128.0f) - mean * mean;
        sMean[row] = mean;
        sRstd[row] = rsqrtf(var + LN_EPS);
    }
    __syncthreads();

    for (int idx = threadIdx.x; idx < 16 * 128; idx += 256) {
        const int row = idx >> 7, col = idx & 127;
        out[(size_t)(m0 + row) * Dd + col] =
            (sT[row][col] - sMean[row]) * sRstd[row] * gamma[col] + beta[col];
    }
}

// ---------------------------------------------------------------------------
extern "C" void kernel_launch(void* const* d_in, const int* in_sizes, int n_in,
                              void* d_out, int out_size, void* d_ws, size_t ws_size,
                              hipStream_t stream) {
    const float* X     = (const float*)d_in[0];
    const float* Wq    = (const float*)d_in[1];
    const float* bq    = (const float*)d_in[2];
    const float* Wk    = (const float*)d_in[3];
    const float* bk    = (const float*)d_in[4];
    const float* Wv    = (const float*)d_in[5];
    const float* bv    = (const float*)d_in[6];
    const float* Wo    = (const float*)d_in[7];
    const float* bo    = (const float*)d_in[8];
    const float* gamma = (const float*)d_in[9];
    const float* beta  = (const float*)d_in[10];
    float* out = (float*)d_out;

    // Workspace layout (bf16 elements)
    __bf16* qt  = (__bf16*)d_ws;                       // [B,H,S,D]  32 MB
    __bf16* kt  = qt  + (size_t)Bb * Hh * Ss * Dd;     // [B,H,S,D]  32 MB
    __bf16* vt  = kt  + (size_t)Bb * Hh * Ss * Dd;     // [B,H,D,S]  32 MB
    __bf16* wv  = vt  + (size_t)Bb * Hh * Dd * Ss;     // [BS, DH]   32 MB
    __bf16* xb  = wv  + (size_t)BS * DH;               // [BS, D]     4 MB
    __bf16* wqt = xb  + (size_t)BS * Dd;               // [DH, D]   256 KB
    __bf16* wkt = wqt + (size_t)DH * Dd;
    __bf16* wvt = wkt + (size_t)DH * Dd;
    __bf16* wot = wvt + (size_t)DH * Dd;               // [D, DH]   256 KB

    cvt_x_kernel<<<dim3((BS * Dd) / 256), 256, 0, stream>>>(X, xb);
    tr_w_kernel<<<dim3((DH * Dd) / 256, 3), 256, 0, stream>>>(Wq, Wk, Wv, wqt, wkt, wvt);
    tr_wo_kernel<<<dim3((Dd * DH) / 256), 256, 0, stream>>>(Wo, wot);

    qkv_kernel<<<dim3(BS / 16, DH / 128, 3), 256, 0, stream>>>(
        xb, wqt, bq, wkt, bk, wvt, bv, qt, kt, vt);
    attn_kernel<<<dim3(Bb * Hh * (Ss / 32) / 8), 256, 0, stream>>>(qt, kt, vt, wv);
    oln_kernel<<<dim3(BS / 16), 256, 0, stream>>>(wv, wot, bo, X, gamma, beta, out);
}
